// Decoder_52845277610164
// MI455X (gfx1250) — matrix operand order repaired and checked by reading, not compiled, not run on hardware
//
#include <hip/hip_runtime.h>

typedef _Float16 h16;
typedef __attribute__((ext_vector_type(16))) _Float16 v16h;
typedef __attribute__((ext_vector_type(8)))  float    v8f;

// ---------------- problem constants ----------------
constexpr int B_ = 256, T_ = 200, H_ = 256, Z_ = 64, DOF_ = 12;
constexpr int XDIM = 96;          // 12 + 64 + 13 padded to 96
constexpr int G4 = 4 * H_;        // 1024
constexpr long BT = (long)B_ * T_;
constexpr long O_MU = 0;
constexpr long O_J  = BT * 42;
constexpr long O_A  = O_J + BT * 12;
constexpr long O_S  = O_A + BT * 12;

union frag16 { v16h v; unsigned int u[8]; };

// A fragment (16xK tile, 16-bit): lane -> M = lane&15; half = lane>>4
// VGPR v holds K = kbase + 16*(v>>2) + 8*half + 2*(v&3) (+1 packed)
__device__ __forceinline__ v16h frag_a(const h16* src, int stride, int rbase,
                                       int kbase, int lane) {
  frag16 f;
  const h16* p = src + (rbase + (lane & 15)) * stride + kbase + ((lane >> 4) << 3);
#pragma unroll
  for (int v = 0; v < 8; ++v)
    f.u[v] = *(const unsigned int*)(p + ((v >> 2) << 4) + ((v & 3) << 1));
  return f.v;
}

// B fragment (Kx16 tile) from row-major weight W[N][K] (so B[k][n] = W[n][k]):
// lane -> N = lane&15; VGPR v holds K = kbase + 16*(v>>2) + 8*half + 2*(v&3) (+1 packed)
__device__ __forceinline__ v16h frag_b(const h16* src, int stride, int nbase,
                                       int kbase, int lane) {
  frag16 f;
  const h16* p = src + (nbase + (lane & 15)) * stride + kbase + ((lane >> 4) << 3);
#pragma unroll
  for (int v = 0; v < 8; ++v)
    f.u[v] = *(const unsigned int*)(p + ((v >> 2) << 4) + ((v & 3) << 1));
  return f.v;
}

__device__ __forceinline__ v8f wmma16(v16h a, v16h b, v8f c) {
  return __builtin_amdgcn_wmma_f32_16x16x32_f16(false, a, false, b, (short)0, c,
                                                false, false);
}

__device__ __forceinline__ float sigmoidf_(float x) {
  return 1.0f / (1.0f + __expf(-x));
}

// fast tanh via single hw exp: tanh(x) = 1 - 2/(exp(2x)+1)
// saturates correctly: x->+inf => 1, x->-inf => -1
__device__ __forceinline__ float tanh_fast(float x) {
  float e = __expf(2.0f * x);
  return 1.0f - 2.0f / (e + 1.0f);
}

// ---------------- prep kernels ----------------
__global__ void pad_cvt_f16(const float* __restrict__ src, h16* __restrict__ dst,
                            int sr, int sc, int dr, int dc) {
  int i = blockIdx.x * blockDim.x + threadIdx.x;
  int n = dr * dc;
  if (i >= n) return;
  int r = i / dc, c = i % dc;
  float v = (r < sr && c < sc) ? src[r * sc + c] : 0.0f;
  dst[i] = (h16)v;
}

__global__ void add_bias(const float* __restrict__ a, const float* __restrict__ b,
                         float* __restrict__ o, int n) {
  int i = blockIdx.x * blockDim.x + threadIdx.x;
  if (i < n) o[i] = a[i] + b[i];
}

// ---------------- persistent LSTM rollout ----------------
__global__ __launch_bounds__(256, 2) void lstm_decoder(
    const h16* __restrict__ Wih16, const h16* __restrict__ Whh16,
    const h16* __restrict__ Wp1h,  const h16* __restrict__ Wp2h,
    const h16* __restrict__ Wo1h,  const h16* __restrict__ Wo2h,
    const h16* __restrict__ Wvh,   const h16* __restrict__ Wfkh,
    const h16* __restrict__ Whzh,  const h16* __restrict__ Wczh,
    const h16* __restrict__ z16,   const float* __restrict__ b4h,
    const float* __restrict__ bhz, const float* __restrict__ bcz,
    const float* __restrict__ bp1, const float* __restrict__ bp2,
    const float* __restrict__ bo1, const float* __restrict__ bo2,
    const float* __restrict__ bv,  const float* __restrict__ bfk,
    const float* __restrict__ pmean, const float* __restrict__ pstd,
    const float* __restrict__ jlow,  const float* __restrict__ jup,
    const float* __restrict__ ddp,   float* __restrict__ out) {
  __shared__ __align__(16) float g_sh[16 * G4];     // 64 KB gates
  __shared__ __align__(16) float c_sh[16 * H_];     // 16 KB cell state
  __shared__ __align__(16) h16   h_sh[16 * H_];     // 8 KB hidden (f16)
  __shared__ __align__(16) h16   x_sh[16 * XDIM];   // 3 KB input row
  __shared__ __align__(16) h16   p1_sh[16 * H_];    // 8 KB
  __shared__ __align__(16) h16   o1_sh[16 * H_];    // 8 KB
  __shared__ __align__(16) h16   cur_sh[16 * 32];   // 1 KB (K padded to 32)
  __shared__ __align__(16) float act_sh[16 * 16];
  __shared__ __align__(16) float obj_sh[16 * 16];
  __shared__ float prev_sh[16 * DOF_];

  const int tid = threadIdx.x;
  const int lane = tid & 31;
  const int wave = tid >> 5;
  const int bbase = blockIdx.x * 16;

  // ---- one-time init ----
  for (int i = tid; i < 16 * XDIM; i += 256) {
    int m = i / XDIM, c = i % XDIM;
    h16 v = (h16)0.0f;
    if (c >= 12 && c < 76) v = z16[(bbase + m) * Z_ + (c - 12)];
    x_sh[i] = v;
  }
  for (int i = tid; i < 16 * 32; i += 256) cur_sh[i] = (h16)0.0f;
  if (tid < 16 * DOF_) prev_sh[tid] = ddp[tid % DOF_];

  // per-thread loop-invariant joint params (threads < 192 use them each step)
  float r_lo = 0.f, r_up = 0.f, r_jm = 0.f, r_jr = 1.f, r_dd = 0.f;
  if (tid < 16 * DOF_) {
    int d = tid % DOF_;
    r_lo = jlow[d];
    r_up = jup[d];
    r_jm = 0.5f * (r_up + r_lo);
    r_jr = 0.5f * (r_up - r_lo);
    r_dd = ddp[d];
  }

  // h0 = z@Whz^T + bhz   (16 tiles over 8 waves, branch-free epilogue)
  for (int tile = wave; tile < 16; tile += 8) {
    const int nb = tile * 16;
    v8f acc = {};
#pragma unroll
    for (int kb = 0; kb < Z_; kb += 32)
      acc = wmma16(frag_a(z16, Z_, bbase, kb, lane),
                   frag_b(Whzh, Z_, nb, kb, lane), acc);
    const int cc = nb + (lane & 15);
    const float bb = bhz[cc];
#pragma unroll
    for (int r = 0; r < 8; ++r) {
      int m = r + ((lane >> 4) << 3);
      h_sh[m * H_ + cc] = (h16)(acc[r] + bb);
    }
  }
  // c0 = z@Wcz^T + bcz   (16 tiles over 8 waves)
  for (int tile = wave; tile < 16; tile += 8) {
    const int nb = tile * 16;
    v8f acc = {};
#pragma unroll
    for (int kb = 0; kb < Z_; kb += 32)
      acc = wmma16(frag_a(z16, Z_, bbase, kb, lane),
                   frag_b(Wczh, Z_, nb, kb, lane), acc);
    const int cc = nb + (lane & 15);
    const float bb = bcz[cc];
#pragma unroll
    for (int r = 0; r < 8; ++r) {
      int m = r + ((lane >> 4) << 3);
      c_sh[m * H_ + cc] = acc[r] + bb;
    }
  }
  __syncthreads();

  // ---- time rollout ----
  for (int t = 0; t < T_; ++t) {
    // S1: refresh prev-norm + time-feature columns of x
    if (tid < 16 * DOF_) {
      int m = tid / DOF_, d = tid % DOF_;
      x_sh[m * XDIM + d] = (h16)((prev_sh[tid] - r_jm) / r_jr);
    }
    if (tid < 13) {
      float tn = (float)t / (float)(T_ - 1);
      float v;
      if (tid == 0) v = tn;
      else {
        int k = (tid - 1) >> 1;
        float ph = 6.283185307179586f * exp2f((float)k) * tn;
        v = (tid & 1) ? sinf(ph) : cosf(ph);
      }
      h16 hv = (h16)v;
      for (int m = 0; m < 16; ++m) x_sh[m * XDIM + 76 + tid] = hv;
    }
    __syncthreads();

    // S2: gates = x@Wih^T + h@Whh^T + (bih+bhh)   (64 tiles, 8/wave)
    for (int tl = 0; tl < 8; ++tl) {
      const int nb = wave * 128 + tl * 16;
      v8f acc = {};
#pragma unroll
      for (int kb = 0; kb < XDIM; kb += 32)
        acc = wmma16(frag_a(x_sh, XDIM, 0, kb, lane),
                     frag_b(Wih16, XDIM, nb, kb, lane), acc);
#pragma unroll
      for (int kb = 0; kb < H_; kb += 32)
        acc = wmma16(frag_a(h_sh, H_, 0, kb, lane),
                     frag_b(Whh16, H_, nb, kb, lane), acc);
      const int cc = nb + (lane & 15);
      const float bb = b4h[cc];
#pragma unroll
      for (int r = 0; r < 8; ++r) {
        int m = r + ((lane >> 4) << 3);
        g_sh[m * G4 + cc] = acc[r] + bb;
      }
    }
    __syncthreads();

    // S3: elementwise LSTM cell update (fast trans-op path)
    for (int u = tid; u < 16 * H_; u += 256) {
      int m = u >> 8, j = u & (H_ - 1);
      float gi = g_sh[m * G4 + j];
      float gf = g_sh[m * G4 + H_ + j];
      float gg = g_sh[m * G4 + 2 * H_ + j];
      float go = g_sh[m * G4 + 3 * H_ + j];
      float cn = sigmoidf_(gf) * c_sh[u] + sigmoidf_(gi) * tanh_fast(gg);
      float hn = sigmoidf_(go) * tanh_fast(cn);
      c_sh[u] = cn;
      h_sh[u] = (h16)hn;
    }
    __syncthreads();

    // S4: p1 = relu(h@Wp1^T+b), o1 = relu(h@Wo1^T+b), sigma head (35 tiles)
    for (int tile = wave; tile < 35; tile += 8) {
      const h16* W;
      int nb;
      int kind;
      if (tile < 16)      { W = Wp1h; nb = tile * 16;        kind = 0; }
      else if (tile < 32) { W = Wo1h; nb = (tile - 16) * 16; kind = 1; }
      else                { W = Wvh;  nb = (tile - 32) * 16; kind = 2; }
      v8f acc = {};
#pragma unroll
      for (int kb = 0; kb < H_; kb += 32)
        acc = wmma16(frag_a(h_sh, H_, 0, kb, lane),
                     frag_b(W, H_, nb, kb, lane), acc);
      const int cc = nb + (lane & 15);
      if (kind == 0) {
        const float bb = bp1[cc];
#pragma unroll
        for (int r = 0; r < 8; ++r) {
          int m = r + ((lane >> 4) << 3);
          p1_sh[m * H_ + cc] = (h16)fmaxf(acc[r] + bb, 0.0f);
        }
      } else if (kind == 1) {
        const float bb = bo1[cc];
#pragma unroll
        for (int r = 0; r < 8; ++r) {
          int m = r + ((lane >> 4) << 3);
          o1_sh[m * H_ + cc] = (h16)fmaxf(acc[r] + bb, 0.0f);
        }
      } else {
        if (cc < 42) {
          const float bb = bv[cc];
#pragma unroll
          for (int r = 0; r < 8; ++r) {
            int m = r + ((lane >> 4) << 3);
            long b = bbase + m;
            float s = sigmoidf_(acc[r] + bb);
            float sg = 0.05f + 0.45f * s;
            out[O_S + (b * T_ + t) * 42 + cc] = __logf(sg);
          }
        }
      }
    }
    __syncthreads();

    // S5: action = tanh(p1@Wp2^T+b) on wave 0; obj = o1@Wo2^T+b on wave 1
    if (wave == 0) {
      v8f acc = {};
#pragma unroll
      for (int kb = 0; kb < H_; kb += 32)
        acc = wmma16(frag_a(p1_sh, H_, 0, kb, lane),
                     frag_b(Wp2h, H_, 0, kb, lane), acc);
      const int cc = lane & 15;
      if (cc < DOF_) {
        const float bb = bp2[cc];
#pragma unroll
        for (int r = 0; r < 8; ++r) {
          int m = r + ((lane >> 4) << 3);
          act_sh[m * 16 + cc] = tanh_fast(acc[r] + bb);
        }
      }
    } else if (wave == 1) {
      v8f acc = {};
#pragma unroll
      for (int kb = 0; kb < H_; kb += 32)
        acc = wmma16(frag_a(o1_sh, H_, 0, kb, lane),
                     frag_b(Wo2h, H_, 0, kb, lane), acc);
      const int cc = lane & 15;
      if (cc < 3) {
        const float bb = bo2[cc];
#pragma unroll
        for (int r = 0; r < 8; ++r) {
          int m = r + ((lane >> 4) << 3);
          obj_sh[m * 16 + cc] = acc[r] + bb;
        }
      }
    }
    __syncthreads();

    // S6: action scaling, straight-through clamp, outputs, obj normalization
    if (tid < 16 * DOF_) {
      int m = tid / DOF_, d = tid % DOF_;
      float a = act_sh[m * 16 + d];
      float target = a * 0.25f + r_dd;
      float cur = prev_sh[tid] + 1.0f * (target - prev_sh[tid]);
      long b = bbase + m;
      long base = b * T_ + t;
      out[O_J + base * 12 + d] = cur;
      out[O_A + base * 12 + d] = a;
      prev_sh[tid] = fminf(fmaxf(cur, r_lo), r_up);
      cur_sh[m * 32 + d] = (h16)cur;   // FK input (unclamped, per reference)
    }
    if (tid < 48) {
      int m = tid / 3, j = tid % 3;
      int cc = 39 + j;
      long b = bbase + m;
      float mu = (obj_sh[m * 16 + j] - pmean[cc]) / pstd[cc];
      out[O_MU + (b * T_ + t) * 42 + cc] = mu;
    }
    __syncthreads();

    // S7: FK = cur@Wfk^T + bfk, normalized, 3 tiles (waves 0..2)
    if (wave < 3) {
      const int nb = wave * 16;
      v8f acc = {};
      acc = wmma16(frag_a(cur_sh, 32, 0, 0, lane),
                   frag_b(Wfkh, 32, nb, 0, lane), acc);
      const int cc = nb + (lane & 15);
      if (cc < 39) {
        const float bb = bfk[cc], pm = pmean[cc], ps = pstd[cc];
#pragma unroll
        for (int r = 0; r < 8; ++r) {
          int m = r + ((lane >> 4) << 3);
          long b = bbase + m;
          out[O_MU + (b * T_ + t) * 42 + cc] = (acc[r] + bb - pm) / ps;
        }
      }
    }
    __syncthreads();
  }
}

// ---------------- host launcher ----------------
extern "C" void kernel_launch(void* const* d_in, const int* in_sizes, int n_in,
                              void* d_out, int out_size, void* d_ws, size_t ws_size,
                              hipStream_t stream) {
  const float* z    = (const float*)d_in[0];
  const float* Whz  = (const float*)d_in[1];
  const float* bhz  = (const float*)d_in[2];
  const float* Wcz  = (const float*)d_in[3];
  const float* bcz  = (const float*)d_in[4];
  const float* Wih  = (const float*)d_in[5];
  const float* Whh  = (const float*)d_in[6];
  const float* bih  = (const float*)d_in[7];
  const float* bhh  = (const float*)d_in[8];
  const float* Wp1  = (const float*)d_in[9];
  const float* bp1  = (const float*)d_in[10];
  const float* Wp2  = (const float*)d_in[11];
  const float* bp2  = (const float*)d_in[12];
  const float* Wo1  = (const float*)d_in[13];
  const float* bo1  = (const float*)d_in[14];
  const float* Wo2  = (const float*)d_in[15];
  const float* bo2  = (const float*)d_in[16];
  const float* Wv   = (const float*)d_in[17];
  const float* bv   = (const float*)d_in[18];
  const float* Wfk  = (const float*)d_in[19];
  const float* bfk  = (const float*)d_in[20];
  const float* pmean = (const float*)d_in[21];
  const float* pstd  = (const float*)d_in[22];
  const float* jlow  = (const float*)d_in[23];
  const float* jup   = (const float*)d_in[24];
  const float* ddp   = (const float*)d_in[25];

  h16* w = (h16*)d_ws;
  size_t off = 0;
  h16* Wih16 = w + off; off += (size_t)1024 * 96;
  h16* Whh16 = w + off; off += (size_t)1024 * 256;
  h16* Wp1h  = w + off; off += (size_t)256 * 256;
  h16* Wp2h  = w + off; off += (size_t)16 * 256;
  h16* Wo1h  = w + off; off += (size_t)256 * 256;
  h16* Wo2h  = w + off; off += (size_t)16 * 256;
  h16* Wvh   = w + off; off += (size_t)48 * 256;
  h16* Wfkh  = w + off; off += (size_t)48 * 32;
  h16* Whzh  = w + off; off += (size_t)256 * 64;
  h16* Wczh  = w + off; off += (size_t)256 * 64;
  h16* z16   = w + off; off += (size_t)256 * 64;
  off = (off + 1) & ~(size_t)1;                 // 4-byte align for floats
  float* b4h = (float*)(w + off);

  auto cvt = [&](const float* s, h16* d, int sr, int sc, int dr, int dc) {
    int n = dr * dc;
    pad_cvt_f16<<<(n + 255) / 256, 256, 0, stream>>>(s, d, sr, sc, dr, dc);
  };
  cvt(Wih, Wih16, 1024, 89, 1024, 96);
  cvt(Whh, Whh16, 1024, 256, 1024, 256);
  cvt(Wp1, Wp1h, 256, 256, 256, 256);
  cvt(Wp2, Wp2h, 12, 256, 16, 256);
  cvt(Wo1, Wo1h, 256, 256, 256, 256);
  cvt(Wo2, Wo2h, 3, 256, 16, 256);
  cvt(Wv,  Wvh,  42, 256, 48, 256);
  cvt(Wfk, Wfkh, 39, 12, 48, 32);
  cvt(Whz, Whzh, 256, 64, 256, 64);
  cvt(Wcz, Wczh, 256, 64, 256, 64);
  cvt(z,   z16,  256, 64, 256, 64);
  add_bias<<<4, 256, 0, stream>>>(bih, bhh, b4h, 1024);

  lstm_decoder<<<B_ / 16, 256, 0, stream>>>(
      Wih16, Whh16, Wp1h, Wp2h, Wo1h, Wo2h, Wvh, Wfkh, Whzh, Wczh, z16, b4h,
      bhz, bcz, bp1, bp2, bo1, bo2, bv, bfk, pmean, pstd, jlow, jup, ddp,
      (float*)d_out);
}
